// HeteroGNN_32152125178509
// MI455X (gfx1250) — compile-verified, matrix operand order checked by reading, not run on hardware
//
#include <hip/hip_runtime.h>
#include <hip/hip_bf16.h>
#include <stdint.h>

// ---------------------------------------------------------------------------
// HeteroGNN (2-layer NNConv, bipartite a<->b), CDNA5 / gfx1250, wave32 WMMA.
//   NA=NB=50000, D=O=64, FE=16, E=30000 per edge type.
// Fused contraction: msg[e,o] = sum_{k,d} h[e,k]*x[e,d]*W2[k,d*64+o]
// as a K=4096(+64 bias) GEMM; A rows are outer products built in registers,
// Wm ([E,64,64] = 491MB/conv) is never materialized. The edge MLP
// h=relu(ea@W1+b1) is fused in-kernel as one K=32 WMMA chunk (bias folded as
// an extra K row). B tiles double-buffered through LDS, 8 waves share each.
// ---------------------------------------------------------------------------

typedef __attribute__((ext_vector_type(16))) __bf16 v16bf;
typedef __attribute__((ext_vector_type(8)))  float  v8f;

#define NA_ 50000
#define NB_ 50000
#define D_  64
#define FE_ 16
#define E_  30000
#define EW_ 8     // waves per edge-conv workgroup

// kd_local within a 32-wide K chunk for (lane-half, element j) in the 16-bit
// A/B operand layout: lanes 0-15 hold K 0..7 (elems 0-7) and 16..23
// (elems 8-15); lanes 16-31 hold K 8..15 and 24..31.
__device__ __forceinline__ int kdloc(int lh, int j) {
  return (lh << 3) + (j < 8 ? j : j + 8);
}

// ---------------------------------------------------------------------------
__global__ void zero_f32(float* __restrict__ p, int n) {
  int i = blockIdx.x * blockDim.x + threadIdx.x;
  int stride = gridDim.x * blockDim.x;
  for (; i < n; i += stride) p[i] = 0.0f;
}

// Pre-layout W2 ([64,4096] f32) + b2 ([4096]) into bf16 WMMA-B tiles:
// 130 chunks (128 for W2, 2 for b2) x 4 o-tiles x 32 lanes x 16 elems.
__global__ void prep_edge_w2(const float* __restrict__ W2,
                             const float* __restrict__ b2,
                             unsigned short* __restrict__ out) {
  int tid = blockIdx.x * blockDim.x + threadIdx.x;
  if (tid >= 130 * 4 * 32 * 16) return;
  int j = tid & 15;
  int l = (tid >> 4) & 31;
  int t = (tid >> 9) & 3;
  int c = tid >> 11;
  int kk = kdloc(l >> 4, j);
  int o  = t * 16 + (l & 15);
  float v;
  if (c < 128) {
    int k = c >> 1;
    int d = ((c & 1) << 5) + kk;
    v = W2[k * 4096 + d * 64 + o];
  } else {
    int d = ((c - 128) << 5) + kk;
    v = b2[d * 64 + o];
  }
  __bf16 bv = (__bf16)v;
  out[tid] = *(unsigned short*)&bv;
}

// Pre-layout W1 ([16,64]) + b1 ([64]) as ONE K=32 chunk x 4 o-tiles:
// rows 0-15 = W1, row 16 = b1 (A supplies 1.0 there), rows 17-31 = 0.
__global__ void prep_edge_w1(const float* __restrict__ W1,
                             const float* __restrict__ b1,
                             unsigned short* __restrict__ out) {
  int tid = blockIdx.x * blockDim.x + threadIdx.x;
  if (tid >= 4 * 32 * 16) return;
  int j = tid & 15;
  int l = (tid >> 4) & 31;
  int t = (tid >> 9) & 3;
  int kk = kdloc(l >> 4, j);
  int o  = t * 16 + (l & 15);
  float v = (kk < 16) ? W1[kk * 64 + o] : (kk == 16 ? b1[o] : 0.0f);
  __bf16 bv = (__bf16)v;
  out[tid] = *(unsigned short*)&bv;
}

// Pre-layout a [64,64] f32 weight into bf16 WMMA-B tiles: 2 chunks x 4 o-tiles.
__global__ void prep_node_w(const float* __restrict__ W,
                            unsigned short* __restrict__ out) {
  int tid = blockIdx.x * blockDim.x + threadIdx.x;
  if (tid >= 2 * 4 * 32 * 16) return;
  int j = tid & 15;
  int l = (tid >> 4) & 31;
  int t = (tid >> 9) & 3;
  int c = tid >> 11;
  int k = c * 32 + kdloc(l >> 4, j);
  int o = t * 16 + (l & 15);
  __bf16 bv = (__bf16)W[k * 64 + o];
  out[tid] = *(unsigned short*)&bv;
}

// ---------------------------------------------------------------------------
// out[n,:] = act( feat[gidx[n]] @ W (bf16 WMMA) + bias [+ msum[n]/max(cnt,1)] )
// One wave per 16-row tile, 4 waves / block. B (8KB) read straight from L2.
__global__ __launch_bounds__(128) void node_gemm(
    const float* __restrict__ feat, const int* __restrict__ gidx,
    const unsigned short* __restrict__ WL, const float* __restrict__ bias,
    const float* __restrict__ msum, const float* __restrict__ mcnt,
    float* __restrict__ out, int N, int do_relu) {
  __shared__ float sx[4][16][64];
  int tid = threadIdx.x, wave = tid >> 5, lane = tid & 31;
  int rbase = blockIdx.x * 64 + wave * 16;

  for (int r = 0; r < 16; ++r) {
    int n = rbase + r;
    float2 v = make_float2(0.0f, 0.0f);
    if (n < N) {
      int src = gidx ? gidx[n] : n;
      v = *(const float2*)(feat + (size_t)src * 64 + lane * 2);
    }
    ((float2*)sx[wave][r])[lane] = v;
  }
  __syncthreads();

  int m = lane & 15, lh = lane >> 4;
  v8f acc[4] = {};
  const v16bf* B = (const v16bf*)WL;
#pragma unroll
  for (int c = 0; c < 2; ++c) {
    v16bf a;
#pragma unroll
    for (int j = 0; j < 16; ++j)
      a[j] = (__bf16)sx[wave][m][c * 32 + kdloc(lh, j)];
#pragma unroll
    for (int t = 0; t < 4; ++t)
      acc[t] = __builtin_amdgcn_wmma_f32_16x16x32_bf16(
          false, a, false, B[(c * 4 + t) * 32 + lane], (short)0, acc[t],
          false, false);
  }

#pragma unroll
  for (int t = 0; t < 4; ++t) {
    int o = t * 16 + m;
    float bo = bias[o];
#pragma unroll
    for (int r = 0; r < 8; ++r) {
      int n = rbase + r + (lh << 3);
      if (n < N) {
        float v = acc[t][r] + bo;
        if (msum) {
          float c = mcnt[n];
          v += msum[(size_t)n * 64 + o] / fmaxf(c, 1.0f);
        }
        if (do_relu) v = fmaxf(v, 0.0f);
        out[(size_t)n * 64 + o] = v;
      }
    }
  }
}

// ---------------------------------------------------------------------------
// Fused edge conv:
//   h[e,:]   = relu(ea[e,:] @ W1 + b1)              (1 WMMA chunk, in LDS)
//   msg[e,o] = sum_{kd} (h[e,k]*x[src[e],d]) * W2L[kd,o]   (130 chunks)
//   atomic scatter (sum + count) by dst.
// One wave per 16-edge tile, 8 waves/WG; W2 B-chunks (4KB) ping-pong
// double-buffered through LDS and shared by all 8 waves.
__global__ __launch_bounds__(32 * EW_) void edge_conv(
    const float* __restrict__ ea,            // [E,16]
    const float* __restrict__ xsrc,          // [Nsrc,64]
    const int* __restrict__ eidx,            // [2,E]: row0=src, row1=dst
    const unsigned short* __restrict__ W1L,  // 1*4*32*16 bf16
    const unsigned short* __restrict__ W2L,  // 130*4*32*16 bf16
    float* __restrict__ ssum, float* __restrict__ cnt, int E) {
  __shared__ float sh_h[EW_][16][64];
  __shared__ float sh_x[EW_][16][64];
  __shared__ float sh_ea[EW_][16][16];
  __shared__ int   sh_d[EW_][16];
  __shared__ v16bf sh_B[2][128];             // ping-pong 4KB B chunks

  int tid = threadIdx.x, wave = tid >> 5, lane = tid & 31;
  int ebase = blockIdx.x * (16 * EW_) + wave * 16;
  int m = lane & 15, lh = lane >> 4;

  // --- load x tile (gather via src index) ---
  for (int r = 0; r < 16; ++r) {
    int e = ebase + r;
    float2 xv = make_float2(0.0f, 0.0f);
    if (e < E) {
      int sn = eidx[e];
      xv = *(const float2*)(xsrc + (size_t)sn * 64 + lane * 2);
    }
    ((float2*)sh_x[wave][r])[lane] = xv;
  }
  // --- load ea tile: 16x16 floats, lane covers 8 ---
  {
    int r = lane >> 1, cb = (lane & 1) * 8;
    int e = ebase + r;
    float4 v0 = make_float4(0, 0, 0, 0), v1 = v0;
    if (e < E) {
      v0 = *(const float4*)(ea + (size_t)e * 16 + cb);
      v1 = *(const float4*)(ea + (size_t)e * 16 + cb + 4);
    }
    *(float4*)&sh_ea[wave][r][cb]     = v0;
    *(float4*)&sh_ea[wave][r][cb + 4] = v1;
  }
  if (lane < 16) {
    int e = ebase + lane;
    sh_d[wave][lane] = (e < E) ? eidx[E + e] : 0;
  }

  // --- fused edge MLP: h = relu(ea @ W1 + b1), one K=32 WMMA chunk ---
  {
    const v16bf* B1 = (const v16bf*)W1L;
    v16bf a;
#pragma unroll
    for (int j = 0; j < 16; ++j) {
      int kk = kdloc(lh, j);
      float v = (kk < 16) ? sh_ea[wave][m][kk] : (kk == 16 ? 1.0f : 0.0f);
      a[j] = (__bf16)v;
    }
#pragma unroll
    for (int t = 0; t < 4; ++t) {
      v8f hacc = {};
      hacc = __builtin_amdgcn_wmma_f32_16x16x32_bf16(
          false, a, false, B1[t * 32 + lane], (short)0, hacc, false, false);
#pragma unroll
      for (int r = 0; r < 8; ++r)
        sh_h[wave][r + (lh << 3)][t * 16 + m] = fmaxf(hacc[r], 0.0f);
    }
  }

  // --- cache this lane's 2x16 x-values (d-halves 0/1) in registers ---
  float xr[2][16];
#pragma unroll
  for (int j = 0; j < 16; ++j) {
    int dl = kdloc(lh, j);
    xr[0][j] = sh_x[wave][m][dl];
    xr[1][j] = sh_x[wave][m][32 + dl];
  }

  v8f acc[4] = {};
  const uint4* Wv = (const uint4*)W2L;

  // preload chunk 0 (256 uint4 = 4KB, 16B per thread)
  ((uint4*)sh_B[0])[tid] = Wv[tid];
  __syncthreads();

  for (int c = 0; c < 130; ++c) {
    int cur = c & 1;
    if (c + 1 < 130)  // stage next chunk into the other buffer
      ((uint4*)sh_B[cur ^ 1])[tid] = Wv[(size_t)(c + 1) * 256 + tid];

    float hk;
    int dh;
    if (c < 128) { hk = sh_h[wave][m][c >> 1]; dh = c & 1; }
    else         { hk = 1.0f;                  dh = c - 128; }  // b2 fold-in

    v16bf a;
#pragma unroll
    for (int j = 0; j < 16; ++j) a[j] = (__bf16)(hk * xr[dh][j]);

#pragma unroll
    for (int t = 0; t < 4; ++t)
      acc[t] = __builtin_amdgcn_wmma_f32_16x16x32_bf16(
          false, a, false, sh_B[cur][t * 32 + lane], (short)0, acc[t],
          false, false);
    __syncthreads();
  }

  // --- degree counts ---
  if (lane < 16) {
    int e = ebase + lane;
    if (e < E) atomicAdd(&cnt[sh_d[wave][lane]], 1.0f);
  }
  // --- scatter msg: C layout -> (edge = r + 8*lh, o = t*16 + m) ---
#pragma unroll
  for (int t = 0; t < 4; ++t) {
    int o = t * 16 + m;
#pragma unroll
    for (int r = 0; r < 8; ++r) {
      int el = r + (lh << 3);
      int e = ebase + el;
      if (e < E) atomicAdd(&ssum[(size_t)sh_d[wave][el] * 64 + o], acc[t][r]);
    }
  }
}

// ---------------------------------------------------------------------------
extern "C" void kernel_launch(void* const* d_in, const int* in_sizes, int n_in,
                              void* d_out, int out_size, void* d_ws,
                              size_t ws_size, hipStream_t stream) {
  (void)in_sizes; (void)n_in; (void)out_size; (void)ws_size;
  const int*   x_a   = (const int*)d_in[0];
  const int*   x_b   = (const int*)d_in[1];
  const int*   ei_ab = (const int*)d_in[2];
  const int*   ei_ba = (const int*)d_in[3];
  const float* ea_ab = (const float*)d_in[4];
  const float* ea_ba = (const float*)d_in[5];
  const float* emb_a = (const float*)d_in[6];
  const float* emb_b = (const float*)d_in[7];
  const float* Wn_a  = (const float*)d_in[8];
  const float* bn_a  = (const float*)d_in[9];
  const float* Wn_b  = (const float*)d_in[10];
  const float* bn_b  = (const float*)d_in[11];
  const float* W1_ab = (const float*)d_in[12];
  const float* b1_ab = (const float*)d_in[13];
  const float* W2_ab = (const float*)d_in[14];
  const float* b2_ab = (const float*)d_in[15];
  const float* W1_ba = (const float*)d_in[16];
  const float* b1_ba = (const float*)d_in[17];
  const float* W2_ba = (const float*)d_in[18];
  const float* b2_ba = (const float*)d_in[19];
  const float* root0_ab = (const float*)d_in[20];
  const float* bias0_ab = (const float*)d_in[21];
  const float* root0_ba = (const float*)d_in[22];
  const float* bias0_ba = (const float*)d_in[23];
  const float* root1_ab = (const float*)d_in[24];
  const float* bias1_ab = (const float*)d_in[25];
  const float* root1_ba = (const float*)d_in[26];
  const float* bias1_ba = (const float*)d_in[27];

  char* ws = (char*)d_ws;
  size_t off = 0;
  const size_t SN  = (size_t)NA_ * 64 * 4;           // node feature buf
  const size_t SW2 = (size_t)130 * 4 * 32 * 16 * 2;  // edge W2 prelayout
  const size_t SW1 = (size_t)4 * 32 * 16 * 2;        // edge W1 prelayout
  const size_t SWN = (size_t)2 * 4 * 32 * 16 * 2;    // node W prelayout
  auto take = [&](size_t sz) {
    size_t p = off; off = (off + sz + 255) & ~(size_t)255; return p;
  };

  float* ha0 = (float*)(ws + take(SN));
  float* hb0 = (float*)(ws + take(SN));
  float* ha1 = (float*)(ws + take(SN));
  float* hb1 = (float*)(ws + take(SN));
  float* ssum = (float*)(ws + take(SN + (size_t)NA_ * 4));  // ssum+cnt contig
  float* cnt  = ssum + (size_t)NA_ * 64;
  unsigned short* W2L_ab = (unsigned short*)(ws + take(SW2));
  unsigned short* W2L_ba = (unsigned short*)(ws + take(SW2));
  unsigned short* W1L_ab = (unsigned short*)(ws + take(SW1));
  unsigned short* W1L_ba = (unsigned short*)(ws + take(SW1));
  unsigned short* WnL_a  = (unsigned short*)(ws + take(SWN));
  unsigned short* WnL_b  = (unsigned short*)(ws + take(SWN));
  unsigned short* rL0ab  = (unsigned short*)(ws + take(SWN));
  unsigned short* rL0ba  = (unsigned short*)(ws + take(SWN));
  unsigned short* rL1ab  = (unsigned short*)(ws + take(SWN));
  unsigned short* rL1ba  = (unsigned short*)(ws + take(SWN));

  const int NODE_GRID  = (NA_ + 63) / 64;
  const int EDGE_GRID  = (E_ + (16 * EW_) - 1) / (16 * EW_);
  const int PREP2_GRID = (130 * 4 * 32 * 16 + 255) / 256;
  const int PREP1_GRID = (4 * 32 * 16 + 255) / 256;
  const int PREPN_GRID = (2 * 4 * 32 * 16 + 255) / 256;

  // Weight pre-layouts (bf16, WMMA-B order)
  prep_edge_w2<<<PREP2_GRID, 256, 0, stream>>>(W2_ab, b2_ab, W2L_ab);
  prep_edge_w2<<<PREP2_GRID, 256, 0, stream>>>(W2_ba, b2_ba, W2L_ba);
  prep_edge_w1<<<PREP1_GRID, 256, 0, stream>>>(W1_ab, b1_ab, W1L_ab);
  prep_edge_w1<<<PREP1_GRID, 256, 0, stream>>>(W1_ba, b1_ba, W1L_ba);
  prep_node_w<<<PREPN_GRID, 256, 0, stream>>>(Wn_a, WnL_a);
  prep_node_w<<<PREPN_GRID, 256, 0, stream>>>(Wn_b, WnL_b);
  prep_node_w<<<PREPN_GRID, 256, 0, stream>>>(root0_ab, rL0ab);
  prep_node_w<<<PREPN_GRID, 256, 0, stream>>>(root0_ba, rL0ba);
  prep_node_w<<<PREPN_GRID, 256, 0, stream>>>(root1_ab, rL1ab);
  prep_node_w<<<PREPN_GRID, 256, 0, stream>>>(root1_ba, rL1ba);

  // Initial node transforms: h = emb[x] @ Wn + bn  (no relu)
  node_gemm<<<NODE_GRID, 128, 0, stream>>>(emb_a, x_a, WnL_a, bn_a,
                                           nullptr, nullptr, ha0, NA_, 0);
  node_gemm<<<NODE_GRID, 128, 0, stream>>>(emb_b, x_b, WnL_b, bn_b,
                                           nullptr, nullptr, hb0, NB_, 0);

  auto run_conv = [&](const float* hsrc, const float* hdst, const int* eidx,
                      const float* ea, const unsigned short* W1L,
                      const unsigned short* W2L, const unsigned short* rootL,
                      const float* bias, float* outbuf, int n_dst) {
    zero_f32<<<512, 256, 0, stream>>>(ssum, n_dst * 64 + n_dst);
    edge_conv<<<EDGE_GRID, 32 * EW_, 0, stream>>>(ea, hsrc, eidx, W1L, W2L,
                                                  ssum, cnt, E_);
    node_gemm<<<(n_dst + 63) / 64, 128, 0, stream>>>(
        hdst, nullptr, rootL, bias, ssum, cnt, outbuf, n_dst, 1);
  };

  // Layer 0 (both convs consume pre-layer ha0/hb0)
  run_conv(ha0, hb0, ei_ab, ea_ab, W1L_ab, W2L_ab, rL0ab, bias0_ab, hb1, NB_);
  run_conv(hb0, ha0, ei_ba, ea_ba, W1L_ba, W2L_ba, rL0ba, bias0_ba, ha1, NA_);

  // Layer 1 -> d_out (ha first, then hb)
  float* out = (float*)d_out;
  run_conv(ha1, hb1, ei_ab, ea_ab, W1L_ab, W2L_ab, rL1ab, bias1_ab,
           out + (size_t)NA_ * 64, NB_);
  run_conv(hb1, ha1, ei_ba, ea_ba, W1L_ba, W2L_ba, rL1ba, bias1_ba,
           out, NA_);
}
